// LSM_Branch_28415503630812
// MI455X (gfx1250) — matrix operand order repaired and checked by reading, not compiled, not run on hardware
//
#include <hip/hip_runtime.h>
#include <hip/hip_bf16.h>

typedef __attribute__((ext_vector_type(16))) _Float16 v16h;
typedef __attribute__((ext_vector_type(8)))  _Float16 v8h;
typedef __attribute__((ext_vector_type(8)))  float    v8f;

#define T_STEPS 32
#define Bb      16
#define Nn      196
#define Cc      384
#define Hh      512
#define Oo      384
#define BNtok   (Bb * Nn)          /* 3136 */
#define ALPHA_SYN 0.9f
#define BETA_MEM  0.8f
#define THRESH    1.0f
#define BN_EPS    1e-5f

// ---------------------------------------------------------------------------
// Fragment loaders (16-bit WMMA operands, contiguous K order k = 16*(lane/16) + s)
// A 16x32: lane holds row (lane&15); 16 contiguous K at koffA
// B 32x16: lane holds col (lane&15); 16 contiguous K at koffB
// ---------------------------------------------------------------------------
__device__ __forceinline__ v16h load_a_frag(const _Float16* __restrict__ row,
                                            int kk, int koffA) {
    v8h lo = *(const v8h*)(row + kk + koffA);
    v8h hi = *(const v8h*)(row + kk + koffA + 8);
    return __builtin_shufflevector(lo, hi,
               0,1,2,3,4,5,6,7,8,9,10,11,12,13,14,15);
}
__device__ __forceinline__ v16h load_b_frag(const _Float16* __restrict__ row,
                                            int kk, int koffB) {
    return *(const v16h*)(row + kk + koffB);
}
__device__ __forceinline__ v8f wmma_f16(v16h a, v16h b, v8f c) {
    return __builtin_amdgcn_wmma_f32_16x16x32_f16(
               false, a, false, b, (short)0, c, false, false);
}

// ---------------------------------------------------------------------------
// f32 -> f16 conversion (weights + input activations)
// ---------------------------------------------------------------------------
__global__ void cvt_f32_to_f16_kernel(const float* __restrict__ src,
                                      _Float16* __restrict__ dst, int n) {
    int i = blockIdx.x * blockDim.x + threadIdx.x;
    if (i < n) dst[i] = (_Float16)src[i];
}

// ---------------------------------------------------------------------------
// Fused per-timestep kernel:  acc = x_t @ W1^T + spk_{t-1} @ Wr^T   (WMMA)
// then RSynaptic state update + Heaviside spike in the epilogue.
// 2x2 register blocking: one wave computes a 32x32 output tile
// (4 WMMAs per 8 b128 loads). 8 waves/block.
// Tiles: (BN/32=98) x (H/32=16) = 1568 waves -> 196 blocks.
// ---------------------------------------------------------------------------
__global__ __launch_bounds__(256)
void lsm_step_kernel(const _Float16* __restrict__ xh_t,     // [BN][C] f16
                     const _Float16* __restrict__ spk_prev, // [BN][H] f16
                     _Float16* __restrict__ spk_next,       // [BN][H] f16
                     float* __restrict__ syn,               // [BN][H] f32
                     float* __restrict__ mem,               // [BN][H] f32
                     const _Float16* __restrict__ W1h,      // [H][C]  f16
                     const _Float16* __restrict__ Wrh,      // [H][H]  f16
                     const float* __restrict__ b1,
                     const float* __restrict__ br) {
    const int lane = threadIdx.x & 31;
    const int wave = threadIdx.x >> 5;
    const int tile = blockIdx.x * 8 + wave;
    const int mt   = tile >> 4;       // 32-row block  (0..97)
    const int nt   = tile & 15;       // 32-col block  (0..15)

    const int arow0 = mt * 32 + (lane & 15);
    const int bcol0 = nt * 32 + (lane & 15);
    const int koffA = (lane & 16) ? 16 : 0;
    const int koffB = (lane & 16) ? 16 : 0;

    v8f acc[2][2] = {};

    // ---- Phase 1: x_t @ W1^T  (K = C = 384) ----
    {
        const _Float16* a0 = xh_t + (size_t)arow0 * Cc;
        const _Float16* a1 = a0 + (size_t)16 * Cc;
        const _Float16* w0 = W1h + (size_t)bcol0 * Cc;
        const _Float16* w1 = w0 + (size_t)16 * Cc;
        for (int kk = 0; kk < Cc; kk += 32) {
            __builtin_prefetch(a0 + kk + 64, 0, 1);
            __builtin_prefetch(a1 + kk + 64, 0, 1);
            v16h fa0 = load_a_frag(a0, kk, koffA);
            v16h fa1 = load_a_frag(a1, kk, koffA);
            v16h fb0 = load_b_frag(w0, kk, koffB);
            v16h fb1 = load_b_frag(w1, kk, koffB);
            acc[0][0] = wmma_f16(fa0, fb0, acc[0][0]);
            acc[0][1] = wmma_f16(fa0, fb1, acc[0][1]);
            acc[1][0] = wmma_f16(fa1, fb0, acc[1][0]);
            acc[1][1] = wmma_f16(fa1, fb1, acc[1][1]);
        }
    }
    // ---- Phase 2: spk_{t-1} @ Wr^T  (K = H = 512) ----
    {
        const _Float16* a0 = spk_prev + (size_t)arow0 * Hh;
        const _Float16* a1 = a0 + (size_t)16 * Hh;
        const _Float16* w0 = Wrh + (size_t)bcol0 * Hh;
        const _Float16* w1 = w0 + (size_t)16 * Hh;
        for (int kk = 0; kk < Hh; kk += 32) {
            __builtin_prefetch(a0 + kk + 64, 0, 1);
            __builtin_prefetch(a1 + kk + 64, 0, 1);
            v16h fa0 = load_a_frag(a0, kk, koffA);
            v16h fa1 = load_a_frag(a1, kk, koffA);
            v16h fb0 = load_b_frag(w0, kk, koffB);
            v16h fb1 = load_b_frag(w1, kk, koffB);
            acc[0][0] = wmma_f16(fa0, fb0, acc[0][0]);
            acc[0][1] = wmma_f16(fa0, fb1, acc[0][1]);
            acc[1][0] = wmma_f16(fa1, fb0, acc[1][0]);
            acc[1][1] = wmma_f16(fa1, fb1, acc[1][1]);
        }
    }

    // ---- Epilogue: RSynaptic update (subtract reset) + spike ----
    #pragma unroll
    for (int ni = 0; ni < 2; ++ni) {
        const int   h    = nt * 32 + ni * 16 + (lane & 15);
        const float bias = b1[h] + br[h];
        #pragma unroll
        for (int mi = 0; mi < 2; ++mi) {
            const int mbase = mt * 32 + mi * 16 + ((lane & 16) ? 8 : 0);
            #pragma unroll
            for (int r = 0; r < 8; ++r) {
                const size_t idx = (size_t)(mbase + r) * Hh + h;
                const float old_mem = mem[idx];
                const float reset   = (old_mem - THRESH) > 0.0f ? 1.0f : 0.0f;
                const float s       = ALPHA_SYN * syn[idx] + acc[mi][ni][r] + bias;
                const float m       = BETA_MEM * old_mem + s - reset * THRESH;
                const float spike   = (m - THRESH) > 0.0f ? 1.0f : 0.0f;
                syn[idx] = s;
                mem[idx] = m;
                spk_next[idx] = (_Float16)spike;
            }
        }
    }
}

// ---------------------------------------------------------------------------
// fc2: out = spk_rec @ W2^T + b2, with fused per-channel sum/sumsq (BN stats).
// 2x2 register blocking. Tiles: (T*BN/32=3136) x (O/32=12) -> 4704 blocks.
// ---------------------------------------------------------------------------
__global__ __launch_bounds__(256)
void fc2_kernel(const _Float16* __restrict__ A,    // [T*BN][H] f16 spikes
                const _Float16* __restrict__ W2h,  // [O][H] f16
                const float* __restrict__ b2,
                float* __restrict__ out,           // [T*BN][O] f32
                float* __restrict__ csum,
                float* __restrict__ csumsq) {
    const int lane = threadIdx.x & 31;
    const int wave = threadIdx.x >> 5;
    const int tile = blockIdx.x * 8 + wave;
    const int NT   = Oo / 32;          // 12
    const int mt   = tile / NT;
    const int nt   = tile - mt * NT;

    const int arow0 = mt * 32 + (lane & 15);
    const int bcol0 = nt * 32 + (lane & 15);
    const int koffA = (lane & 16) ? 16 : 0;
    const int koffB = (lane & 16) ? 16 : 0;

    v8f acc[2][2] = {};
    const _Float16* a0 = A + (size_t)arow0 * Hh;
    const _Float16* a1 = a0 + (size_t)16 * Hh;
    const _Float16* w0 = W2h + (size_t)bcol0 * Hh;
    const _Float16* w1 = w0 + (size_t)16 * Hh;
    for (int kk = 0; kk < Hh; kk += 32) {
        __builtin_prefetch(a0 + kk + 64, 0, 1);
        __builtin_prefetch(a1 + kk + 64, 0, 1);
        v16h fa0 = load_a_frag(a0, kk, koffA);
        v16h fa1 = load_a_frag(a1, kk, koffA);
        v16h fb0 = load_b_frag(w0, kk, koffB);
        v16h fb1 = load_b_frag(w1, kk, koffB);
        acc[0][0] = wmma_f16(fa0, fb0, acc[0][0]);
        acc[0][1] = wmma_f16(fa0, fb1, acc[0][1]);
        acc[1][0] = wmma_f16(fa1, fb0, acc[1][0]);
        acc[1][1] = wmma_f16(fa1, fb1, acc[1][1]);
    }

    #pragma unroll
    for (int ni = 0; ni < 2; ++ni) {
        const int   n    = nt * 32 + ni * 16 + (lane & 15);
        const float bias = b2[n];
        float ps = 0.0f, pq = 0.0f;
        #pragma unroll
        for (int mi = 0; mi < 2; ++mi) {
            const int mbase = mt * 32 + mi * 16 + ((lane & 16) ? 8 : 0);
            #pragma unroll
            for (int r = 0; r < 8; ++r) {
                const float val = acc[mi][ni][r] + bias;
                out[(size_t)(mbase + r) * Oo + n] = val;
                ps += val;
                pq += val * val;
            }
        }
        // lanes l and l+16 share channel n: fold before atomics
        ps += __shfl_xor(ps, 16, 32);
        pq += __shfl_xor(pq, 16, 32);
        if (lane < 16) {
            atomicAdd(&csum[n], ps);
            atomicAdd(&csumsq[n], pq);
        }
    }
}

// ---------------------------------------------------------------------------
// BatchNorm finalize: per-channel scale/shift from batch stats
// ---------------------------------------------------------------------------
__global__ void bn_finalize_kernel(const float* __restrict__ csum,
                                   const float* __restrict__ csumsq,
                                   const float* __restrict__ gamma,
                                   const float* __restrict__ beta,
                                   float* __restrict__ scale,
                                   float* __restrict__ shift) {
    int o = blockIdx.x * blockDim.x + threadIdx.x;
    if (o < Oo) {
        const float inv_n = 1.0f / (float)((size_t)T_STEPS * BNtok);
        const float mu    = csum[o] * inv_n;
        const float var   = csumsq[o] * inv_n - mu * mu;
        const float rstd  = rsqrtf(var + BN_EPS);
        const float sc    = rstd * gamma[o];
        scale[o] = sc;
        shift[o] = beta[o] - mu * sc;
    }
}

// ---------------------------------------------------------------------------
// ParametricLIF scan over T (hard reset), fused with BN normalize
// ---------------------------------------------------------------------------
__global__ void lif_kernel(const float* __restrict__ out,   // [T][BN][O]
                           const float* __restrict__ scale,
                           const float* __restrict__ shift,
                           const float* __restrict__ plif_w,
                           float* __restrict__ spikes) {    // [T][BN][O]
    const int idx = blockIdx.x * blockDim.x + threadIdx.x;
    const int per_t = BNtok * Oo;                 // 1,204,224
    if (idx >= per_t) return;
    const int   o   = idx % Oo;
    const float sc  = scale[o];
    const float sh  = shift[o];
    const float sig = 1.0f / (1.0f + __expf(-plif_w[0]));  // sigmoid(plif_w)
    float v = 0.0f;
    for (int t = 0; t < T_STEPS; ++t) {
        const size_t off = (size_t)t * per_t + idx;
        const float u = out[off] * sc + sh;
        v = v + (u - v) * sig;
        const float s = (v - 1.0f) > 0.0f ? 1.0f : 0.0f;
        spikes[off] = s;
        v = (1.0f - s) * v;   // hard reset, detach
    }
}

// ---------------------------------------------------------------------------
// Host launcher
// ---------------------------------------------------------------------------
extern "C" void kernel_launch(void* const* d_in, const int* in_sizes, int n_in,
                              void* d_out, int out_size, void* d_ws, size_t ws_size,
                              hipStream_t stream) {
    const float* x    = (const float*)d_in[0];   // [T,B,N,C]
    const float* W1   = (const float*)d_in[1];   // [H,C]
    const float* b1   = (const float*)d_in[2];   // [H]
    const float* Wr   = (const float*)d_in[3];   // [H,H]
    const float* br   = (const float*)d_in[4];   // [H]
    const float* W2   = (const float*)d_in[5];   // [O,H]
    const float* b2   = (const float*)d_in[6];   // [O]
    const float* gam  = (const float*)d_in[7];   // [O]
    const float* bet  = (const float*)d_in[8];   // [O]
    const float* plw  = (const float*)d_in[9];   // [1]
    float* spikes = (float*)d_out;

    char* ws = (char*)d_ws;
    size_t off = 0;
    auto take = [&](size_t bytes) -> void* {
        void* p = ws + off;
        off = (off + bytes + 255) & ~(size_t)255;
        return p;
    };

    const size_t nx = (size_t)T_STEPS * BNtok * Cc;        // 38,535,168
    _Float16* xh      = (_Float16*)take(nx * 2);
    _Float16* W1h     = (_Float16*)take((size_t)Hh * Cc * 2);
    _Float16* Wrh     = (_Float16*)take((size_t)Hh * Hh * 2);
    _Float16* W2h     = (_Float16*)take((size_t)Oo * Hh * 2);
    _Float16* spk_buf = (_Float16*)take((size_t)(T_STEPS + 1) * BNtok * Hh * 2);
    float*    syn     = (float*)take((size_t)BNtok * Hh * 4);
    float*    mem     = (float*)take((size_t)BNtok * Hh * 4);
    float*    outbuf  = (float*)take((size_t)T_STEPS * BNtok * Oo * 4);
    float*    csum    = (float*)take(Oo * 4);
    float*    csumsq  = (float*)take(Oo * 4);
    float*    scale   = (float*)take(Oo * 4);
    float*    shift   = (float*)take(Oo * 4);

    // Zero-init states (graph-capturable async memsets)
    hipMemsetAsync(spk_buf, 0, (size_t)BNtok * Hh * 2, stream);  // spk at t=-1
    hipMemsetAsync(syn,     0, (size_t)BNtok * Hh * 4, stream);
    hipMemsetAsync(mem,     0, (size_t)BNtok * Hh * 4, stream);
    hipMemsetAsync(csum,    0, Oo * 4, stream);
    hipMemsetAsync(csumsq,  0, Oo * 4, stream);

    // Convert activations + weights to f16 (spikes are {0,1}: lossless A-operand)
    cvt_f32_to_f16_kernel<<<(int)((nx + 255) / 256), 256, 0, stream>>>(x, xh, (int)nx);
    cvt_f32_to_f16_kernel<<<(Hh * Cc + 255) / 256, 256, 0, stream>>>(W1, W1h, Hh * Cc);
    cvt_f32_to_f16_kernel<<<(Hh * Hh + 255) / 256, 256, 0, stream>>>(Wr, Wrh, Hh * Hh);
    cvt_f32_to_f16_kernel<<<(Oo * Hh + 255) / 256, 256, 0, stream>>>(W2, W2h, Oo * Hh);

    // Recurrent scan: fused (fc1 + spk@Wr^T) WMMA GEMM + state update per step
    const int step_blocks = (BNtok / 32) * (Hh / 32) / 8;   // 1568 waves / 8 = 196
    for (int t = 0; t < T_STEPS; ++t) {
        lsm_step_kernel<<<step_blocks, 256, 0, stream>>>(
            xh + (size_t)t * BNtok * Cc,
            spk_buf + (size_t)t * BNtok * Hh,
            spk_buf + (size_t)(t + 1) * BNtok * Hh,
            syn, mem, W1h, Wrh, b1, br);
    }

    // fc2 + BN stats
    const int fc2_blocks = ((T_STEPS * BNtok / 32) * (Oo / 32)) / 8;  // 4704
    fc2_kernel<<<fc2_blocks, 256, 0, stream>>>(
        spk_buf + (size_t)BNtok * Hh,   // spikes for t=0..T-1
        W2h, b2, outbuf, csum, csumsq);

    bn_finalize_kernel<<<(Oo + 255) / 256, 256, 0, stream>>>(
        csum, csumsq, gam, bet, scale, shift);

    // BN normalize + PLIF scan -> final spikes
    const int per_t = BNtok * Oo;
    lif_kernel<<<(per_t + 255) / 256, 256, 0, stream>>>(
        outbuf, scale, shift, plw, spikes);
}
